// Modal_encoders_32925219291189
// MI455X (gfx1250) — compile-verified, matrix operand order checked by reading, not run on hardware
//
#include <hip/hip_runtime.h>
#include <hip/hip_bf16.h>

typedef __attribute__((ext_vector_type(16))) _Float16 v16h;
typedef __attribute__((ext_vector_type(8)))  float    v8f;

#define B_SZ 16
#define NPB  16384            // tokens per batch (128*128)
#define NTOK (B_SZ*NPB)       // 262144
#define SM   96
#define EPSV 1e-5f

// ---------------- K0: weight prep (transpose + fp16) ----------------
__global__ void k0_prep(const float* __restrict__ exp_w, const float* __restrict__ out_w,
                        _Float16* __restrict__ ewT, _Float16* __restrict__ outwT)
{
    int i = blockIdx.x * 256 + threadIdx.x;
    if (i < 3*4*32*32) {                       // ewT[be][d][c] = exp_w[be][c][d]
        int c = i & 31, d = (i >> 5) & 31, be = i >> 10;
        ewT[(be*32 + d)*32 + c] = (_Float16)exp_w[(be*32 + c)*32 + d];
    }
    int j = i - 3*4*32*32;
    if (j >= 0 && j < 64*96) {                 // outwT[f][k] = out_w[k][f]
        int k = j % 96, f = j / 96;
        outwT[f*96 + k] = (_Float16)out_w[k*64 + f];
    }
}

// ---------------- K1: patch conv + bias + BN partial stats ----------------
// block = 256 threads; covers (branch, batch, out-row h, 64-token w segment)
__global__ void k1_conv(const float* __restrict__ x0, const float* __restrict__ x1,
                        const float* __restrict__ x2,
                        const float* __restrict__ conv_w, const float* __restrict__ conv_b,
                        float* __restrict__ Xraw, float* __restrict__ sums, float* __restrict__ sumsq)
{
    __shared__ float xl[3*4*256];   // [c][r][256 input cols]
    __shared__ float wl[32*48];     // [dout][c*16+r*4+s]
    __shared__ float red[256];
    int t = threadIdx.x;
    int h = blockIdx.x >> 1, wseg = blockIdx.x & 1;
    int b = blockIdx.y, br = blockIdx.z;
    const float* xp = (br == 0) ? x0 : (br == 1 ? x1 : x2);
    size_t ib = (size_t)b * 3u * 262144u;
    for (int i = t; i < 3072; i += 256) {
        int c = i >> 10, rem = i & 1023, rr = rem >> 8, col = rem & 255;
        xl[i] = xp[ib + (size_t)c*262144 + (size_t)(4*h + rr)*512 + wseg*256 + col];
    }
    for (int i = t; i < 1536; i += 256) wl[i] = conv_w[br*1536 + i];
    __syncthreads();

    int dout = t & 31, tg = t >> 5;
    float cb = conv_b[br*32 + dout];
    float s1 = 0.f, s2 = 0.f;
    for (int k = 0; k < 8; ++k) {
        int wloc = tg*8 + k;                    // 0..63 within segment
        float o = cb;
        const float* wp = &wl[dout*48];
        const float* xb = &xl[wloc*4];
#pragma unroll
        for (int c = 0; c < 3; ++c)
#pragma unroll
            for (int rr = 0; rr < 4; ++rr)
#pragma unroll
                for (int s = 0; s < 4; ++s)
                    o += xb[c*1024 + rr*256 + s] * wp[c*16 + rr*4 + s];
        int n = h*128 + wseg*64 + wloc;
        Xraw[((size_t)b*NPB + n)*SM + br*32 + dout] = o;
        s1 += o; s2 += o*o;
    }
    red[t] = s1; __syncthreads();
    if (t < 32) { float a = 0; for (int g = 0; g < 8; ++g) a += red[t + 32*g];
                  atomicAdd(&sums[br*32 + t], a); }
    __syncthreads();
    red[t] = s2; __syncthreads();
    if (t < 32) { float a = 0; for (int g = 0; g < 8; ++g) a += red[t + 32*g];
                  atomicAdd(&sumsq[br*32 + t], a); }
}

// ---------------- K2: BN finalize to per-channel affine ----------------
__global__ void k2_bn(const float* sums, const float* sumsq, const float* bn_g,
                      const float* bn_b, float* bnA, float* bnB)
{
    int c = threadIdx.x;
    if (c < 96) {
        const float cnt = (float)NTOK;
        float mu  = sums[c] / cnt;
        float var = sumsq[c] / cnt - mu*mu;
        float a   = bn_g[c] * rsqrtf(var + EPSV);
        bnA[c] = a;  bnB[c] = bn_b[c] - mu*a;
    }
}

// ---------------- K3: BN apply + top-2 MoE via WMMA, in-place ----------------
// block = 256 = 8 waves; wave handles 16 tokens of one branch
__global__ void k3_moe(float* __restrict__ X, const float* __restrict__ gate_w,
                       const float* __restrict__ exp_b, const _Float16* __restrict__ ewT,
                       const float* __restrict__ bnA, const float* __restrict__ bnB)
{
    __shared__ float tile[8][16][32];
    __shared__ float wf[8][16][4];
    int wv = threadIdx.x >> 5, lane = threadIdx.x & 31;
    int br = blockIdx.y;
    int tok0 = (blockIdx.x * 8 + wv) * 16;
    int r = lane & 15, half = lane >> 4, kh = half * 8;

    {   // load 16 channels per lane, apply BN affine
        const float* src = X + (size_t)(tok0 + r)*SM + br*32 + half*16;
#pragma unroll
        for (int j = 0; j < 16; ++j) {
            int c = half*16 + j;
            tile[wv][r][c] = src[j]*bnA[br*32 + c] + bnB[br*32 + c];
        }
    }
    __syncthreads();
    if (lane < 16) {            // gate + top-2 softmax (jax tie-break: lower index)
        float lg[4] = {0,0,0,0};
#pragma unroll
        for (int c = 0; c < 32; ++c) {
            float xv = tile[wv][lane][c];
            const float* gp = &gate_w[(br*32 + c)*4];
#pragma unroll
            for (int e = 0; e < 4; ++e) lg[e] += xv*gp[e];
        }
        int i0 = 0;
#pragma unroll
        for (int e = 1; e < 4; ++e) if (lg[e] > lg[i0]) i0 = e;
        int i1 = -1;
#pragma unroll
        for (int e = 0; e < 4; ++e) if (e != i0 && (i1 < 0 || lg[e] > lg[i1])) i1 = e;
        float w0 = 1.f / (1.f + __expf(lg[i1] - lg[i0]));
        float w1 = 1.f - w0;
#pragma unroll
        for (int e = 0; e < 4; ++e) wf[wv][lane][e] = (e == i0) ? w0 : ((e == i1) ? w1 : 0.f);
    }
    __syncthreads();

    v16h a;                     // A frag: row r, K-half kh
#pragma unroll
    for (int j = 0; j < 8; ++j) {
        a[j]     = (_Float16)tile[wv][r][kh + j];
        a[j + 8] = (_Float16)tile[wv][r][kh + 16 + j];
    }
    v8f acc0 = {}; v8f acc1 = {};
#pragma unroll
    for (int e = 0; e < 4; ++e) {
#pragma unroll
        for (int nh = 0; nh < 2; ++nh) {
            int col = nh*16 + r;
            const _Float16* bp = ewT + ((size_t)(br*4 + e)*32 + col)*32 + kh;
            v16h bb;
#pragma unroll
            for (int j = 0; j < 8; ++j) { bb[j] = bp[j]; bb[j + 8] = bp[16 + j]; }
            v8f cz = {};
            cz = __builtin_amdgcn_wmma_f32_16x16x32_f16(false, a, false, bb, (short)0, cz, false, false);
            float ebv = exp_b[(br*4 + e)*32 + col];
#pragma unroll
            for (int v = 0; v < 8; ++v) {
                float wt = wf[wv][v + half*8][e];
                float val = wt * (cz[v] + ebv);
                if (nh == 0) acc0[v] += val; else acc1[v] += val;
            }
        }
    }
#pragma unroll
    for (int v = 0; v < 8; ++v) {           // in-place write of this branch's slice
        size_t row = (size_t)(tok0 + v + half*8)*SM + br*32;
        X[row + r]      = acc0[v];
        X[row + 16 + r] = acc1[v];
    }
}

// ---------------- K4: Soft-MoE logits + dispatch numerators ----------------
__global__ void k4_logits(const float* __restrict__ X, const float* __restrict__ phi,
                          float* __restrict__ L, float* __restrict__ sumexp,
                          float* __restrict__ slotsNum)
{
    __shared__ float exL[256*6];
    int t = threadIdx.x, b = blockIdx.y, n0 = blockIdx.x * 256, n = n0 + t;
    const float* xp = X + ((size_t)b*NPB + n)*SM;
    float lg[6] = {0,0,0,0,0,0};
    for (int d = 0; d < 96; ++d) {
        float xv = xp[d];
        const float* pp = &phi[d*6];
#pragma unroll
        for (int es = 0; es < 6; ++es) lg[es] += xv*pp[es];
    }
    float* lp = L + ((size_t)b*NPB + n)*6;
#pragma unroll
    for (int es = 0; es < 6; ++es) { lp[es] = lg[es]; exL[t*6 + es] = __expf(lg[es]); }
    __syncthreads();
    for (int p = t; p < 576; p += 256) {
        int es = p / 96, d = p % 96;
        float s = 0.f;
        for (int k = 0; k < 256; ++k)
            s += exL[k*6 + es] * X[((size_t)b*NPB + n0 + k)*SM + d];
        atomicAdd(&slotsNum[(b*6 + es)*96 + d], s);
    }
    if (t < 6) {
        float s = 0.f;
        for (int k = 0; k < 256; ++k) s += exL[k*6 + t];
        atomicAdd(&sumexp[b*6 + t], s);
    }
}

// ---------------- K5: per-slot expert linear ----------------
__global__ void k5_yslots(const float* slotsNum, const float* sumexp,
                          const float* smoe_w, const float* smoe_b, float* yslots)
{
    int bes = blockIdx.x;            // b*6 + es
    int es = bes % 6, e = es >> 1;
    int f = threadIdx.x;             // 0..95
    float inv = 1.f / sumexp[bes];
    const float* sp = &slotsNum[bes*96];
    const float* wp = &smoe_w[e*96*96];
    float acc = smoe_b[e*96 + f];
    for (int d = 0; d < 96; ++d) acc += (sp[d]*inv) * wp[d*96 + f];
    yslots[bes*96 + f] = acc;
}

// ---------------- K6: combine + out projection via WMMA ----------------
// block = 256 = 8 waves, 128 tokens of one batch
__global__ void k6_out(const float* __restrict__ L, const float* __restrict__ yslots,
                       const _Float16* __restrict__ outwT, const float* __restrict__ out_b,
                       float* __restrict__ out)
{
    __shared__ float ysl[6*96];
    __shared__ float Yt[8][16][96];
    int t = threadIdx.x, b = blockIdx.y;
    int wv = t >> 5, lane = t & 31;
    int r = lane & 15, dh = lane >> 4, kh = dh * 8;
    int n0 = blockIdx.x * 128 + wv * 16;
    for (int i = t; i < 576; i += 256) ysl[i] = yslots[b*576 + i];
    __syncthreads();
    {   // combine softmax over 6, Y = combine @ yslots (each lane: 48 dims)
        const float* lp = L + ((size_t)b*NPB + n0 + r)*6;
        float lg[6], m = -1e30f;
#pragma unroll
        for (int es = 0; es < 6; ++es) { lg[es] = lp[es]; m = fmaxf(m, lg[es]); }
        float ex[6], s = 0.f;
#pragma unroll
        for (int es = 0; es < 6; ++es) { ex[es] = __expf(lg[es] - m); s += ex[es]; }
        float inv = 1.f / s;
        for (int d = dh*48; d < dh*48 + 48; ++d) {
            float y = 0.f;
#pragma unroll
            for (int es = 0; es < 6; ++es) y += ex[es]*inv * ysl[es*96 + d];
            Yt[wv][r][d] = y;
        }
    }
    __syncthreads();
    v16h a[3];
#pragma unroll
    for (int s = 0; s < 3; ++s)
#pragma unroll
        for (int j = 0; j < 8; ++j) {
            a[s][j]     = (_Float16)Yt[wv][r][s*32 + kh + j];
            a[s][j + 8] = (_Float16)Yt[wv][r][s*32 + kh + 16 + j];
        }
#pragma unroll
    for (int nh = 0; nh < 4; ++nh) {
        int f = nh*16 + r;
        v8f acc = {};
#pragma unroll
        for (int s = 0; s < 3; ++s) {
            const _Float16* bp = outwT + f*96 + s*32 + kh;
            v16h bb;
#pragma unroll
            for (int j = 0; j < 8; ++j) { bb[j] = bp[j]; bb[j + 8] = bp[16 + j]; }
            acc = __builtin_amdgcn_wmma_f32_16x16x32_f16(false, a[s], false, bb, (short)0, acc, false, false);
        }
        float ob = out_b[f];
        float* dst = out + ((size_t)b*64 + f)*NPB + n0 + dh*8;
        float4 lo = make_float4(acc[0]+ob, acc[1]+ob, acc[2]+ob, acc[3]+ob);
        float4 hi = make_float4(acc[4]+ob, acc[5]+ob, acc[6]+ob, acc[7]+ob);
        *(float4*)(dst)     = lo;
        *(float4*)(dst + 4) = hi;
    }
}

extern "C" void kernel_launch(void* const* d_in, const int* in_sizes, int n_in,
                              void* d_out, int out_size, void* d_ws, size_t ws_size,
                              hipStream_t stream)
{
    (void)in_sizes; (void)n_in; (void)out_size; (void)ws_size;
    const float* x0     = (const float*)d_in[0];
    const float* x1     = (const float*)d_in[1];
    const float* x2     = (const float*)d_in[2];
    const float* conv_w = (const float*)d_in[3];
    const float* conv_b = (const float*)d_in[4];
    const float* bn_g   = (const float*)d_in[5];
    const float* bn_b   = (const float*)d_in[6];
    const float* gate_w = (const float*)d_in[7];
    const float* exp_w  = (const float*)d_in[8];
    const float* exp_b  = (const float*)d_in[9];
    const float* phi    = (const float*)d_in[10];
    const float* smoe_w = (const float*)d_in[11];
    const float* smoe_b = (const float*)d_in[12];
    const float* out_w  = (const float*)d_in[13];
    const float* out_b  = (const float*)d_in[14];

    char* ws = (char*)d_ws;
    float*    Xraw     = (float*)(ws);                          // 100,663,296 B
    float*    Lbuf     = (float*)(ws + 100663296ull);           //   6,291,456 B
    float*    acc      = (float*)(ws + 106954752ull);           // atomics region
    float*    sums     = acc;
    float*    sumsq    = acc + 96;
    float*    sumexp   = acc + 192;
    float*    slotsNum = acc + 288;                             // 9216 floats
    float*    bnA      = (float*)(ws + 106992768ull);
    float*    bnB      = bnA + 96;
    float*    yslots   = (float*)(ws + 106993536ull);           // 36,864 B
    _Float16* ewT      = (_Float16*)(ws + 107030400ull);        // 24,576 B
    _Float16* outwT    = (_Float16*)(ws + 107054976ull);        // 12,288 B

    hipMemsetAsync(acc, 0, (size_t)(288 + 9216)*sizeof(float), stream);
    k0_prep  <<<72, 256, 0, stream>>>(exp_w, out_w, ewT, outwT);
    k1_conv  <<<dim3(256, 16, 3), 256, 0, stream>>>(x0, x1, x2, conv_w, conv_b, Xraw, sums, sumsq);
    k2_bn    <<<1, 96, 0, stream>>>(sums, sumsq, bn_g, bn_b, bnA, bnB);
    k3_moe   <<<dim3(2048, 3), 256, 0, stream>>>(Xraw, gate_w, exp_b, ewT, bnA, bnB);
    k4_logits<<<dim3(64, 16), 256, 0, stream>>>(Xraw, phi, Lbuf, sumexp, slotsNum);
    k5_yslots<<<96, 96, 0, stream>>>(slotsNum, sumexp, smoe_w, smoe_b, yslots);
    k6_out   <<<dim3(128, 16), 256, 0, stream>>>(Lbuf, yslots, outwT, out_b, (float*)d_out);
}
